// GraphEncoder_56006373540170
// MI455X (gfx1250) — matrix operand order repaired and checked by reading, not compiled, not run on hardware
//
#include <hip/hip_runtime.h>

#define NN 50000
#define EE 312500
#define GG 50
#define IN_DIM 384
#define HD 256

#define F_RELU  1
#define F_RESID 2

typedef __attribute__((ext_vector_type(16))) _Float16 v16h;
typedef __attribute__((ext_vector_type(8)))  float    v8f;

// ---------------------------------------------------------------------------
// Pack fp32 row-major weight W[K x 256] into f16 WMMA B-fragment layout.
// Tile (kt, nt) is 32(K) x 16(N). Per lane: n = nt*16 + (lane&15),
// ksel = lane>>4; dword d holds halves for k = kt*32 + ksel*8 + 2d (+8 if d>=4), +1.
// Storage: Wp[((kt*16 + nt)*32 + lane)*8 + d]  -> contiguous 32B per lane.
// ---------------------------------------------------------------------------
__global__ void pack_w_kernel(const float* __restrict__ W,
                              unsigned int* __restrict__ Wp, int K) {
    int idx = blockIdx.x * blockDim.x + threadIdx.x;
    int ktiles = K >> 5;
    if (idx >= ktiles * 512) return;           // 16 ntiles * 32 lanes = 512
    int lane = idx & 31;
    int nt   = (idx >> 5) & 15;
    int kt   = idx >> 9;
    int n    = nt * 16 + (lane & 15);
    int ksel = lane >> 4;
    unsigned int* dst = Wp + ((size_t)(kt * 16 + nt) * 32 + lane) * 8;
#pragma unroll
    for (int d = 0; d < 8; ++d) {
        int k0 = kt * 32 + ksel * 8 + 2 * d + (d >= 4 ? 8 : 0);
        union { _Float16 h[2]; unsigned int u; } cv;
        cv.h[0] = (_Float16)W[(size_t)k0 * HD + n];
        cv.h[1] = (_Float16)W[(size_t)(k0 + 1) * HD + n];
        dst[d] = cv.u;
    }
}

// ---------------------------------------------------------------------------
// C[M x 256] = epilogue( A[M x K] @ Wpacked + bias )
// Block: 256 threads = 8 waves; block tile 32 x 256; wave tile 16 x 64.
// A fragment (16x32 f16) per ISA layout: lane group 0 -> K 0-7 & 16-23,
// group 1 -> K 8-15 & 24-31 (halves 0..7 then 8..15).
// ---------------------------------------------------------------------------
__global__ __launch_bounds__(256)
void wmma_gemm_kernel(const float* __restrict__ A,
                      const unsigned int* __restrict__ Wp,
                      const float* __restrict__ bias,
                      const float* __restrict__ resid,
                      float* __restrict__ C,
                      int M, int K, int flags) {
    const int lane  = threadIdx.x & 31;
    const int wave  = threadIdx.x >> 5;
    const int mlane = lane & 15;
    const int ksel  = lane >> 4;
    const int row0  = blockIdx.x * 32 + (wave >> 2) * 16;
    const int nt0   = (wave & 3) * 4;           // 4 n-tiles of 16 per wave
    const int row   = row0 + mlane;
    const bool rowok = row < M;

    v8f acc[4] = {};
    const int ktiles = K >> 5;
    for (int kt = 0; kt < ktiles; ++kt) {
        float fa[16];
        if (rowok) {
            const float* ap = A + (size_t)row * K + kt * 32 + ksel * 8;
            const float4 f0 = *(const float4*)(ap);
            const float4 f1 = *(const float4*)(ap + 4);
            const float4 f2 = *(const float4*)(ap + 16);
            const float4 f3 = *(const float4*)(ap + 20);
            fa[0]=f0.x;  fa[1]=f0.y;  fa[2]=f0.z;  fa[3]=f0.w;
            fa[4]=f1.x;  fa[5]=f1.y;  fa[6]=f1.z;  fa[7]=f1.w;
            fa[8]=f2.x;  fa[9]=f2.y;  fa[10]=f2.z; fa[11]=f2.w;
            fa[12]=f3.x; fa[13]=f3.y; fa[14]=f3.z; fa[15]=f3.w;
        } else {
#pragma unroll
            for (int i = 0; i < 16; ++i) fa[i] = 0.f;
        }
        v16h a;
#pragma unroll
        for (int i = 0; i < 16; ++i) a[i] = (_Float16)fa[i];

        const unsigned int* wb =
            Wp + ((size_t)(kt * 16 + nt0) * 32 + lane) * 8;
#pragma unroll
        for (int j = 0; j < 4; ++j) {
            v16h b = *(const v16h*)(wb + (size_t)j * 256); // next n-tile: 32 lanes * 8 dw
            acc[j] = __builtin_amdgcn_wmma_f32_16x16x32_f16(
                false, a, false, b, (short)0, acc[j], false, false);
        }
    }

    // C layout: VGPR r -> local row (ksel*8 + r), col = n-tile base + mlane
#pragma unroll
    for (int j = 0; j < 4; ++j) {
        const int n  = (nt0 + j) * 16 + mlane;
        const float bv = bias[n];
#pragma unroll
        for (int r = 0; r < 8; ++r) {
            const int rm = row0 + ksel * 8 + r;
            if (rm < M) {
                float v = acc[j][r] + bv;
                if (flags & F_RESID) v += resid[(size_t)rm * HD + n];
                if (flags & F_RELU)  v = fmaxf(v, 0.f);
                C[(size_t)rm * HD + n] = v;
            }
        }
    }
}

// z[i] = (1 + eps) * h[i]   (aggr initialized to self-term; atomics add messages)
__global__ void scale_copy_kernel(const float* __restrict__ h,
                                  const float* __restrict__ epsPtr,
                                  float* __restrict__ z, int n) {
    int i = blockIdx.x * blockDim.x + threadIdx.x;
    if (i < n) z[i] = (1.0f + *epsPtr) * h[i];
}

// per edge: aggr[dst] += relu(h[src] + e)   (float4 per thread, float atomics)
__global__ void msg_scatter_kernel(const float* __restrict__ h,
                                   const float* __restrict__ e,
                                   const long long* __restrict__ ei,
                                   float* __restrict__ aggr) {
    int idx = blockIdx.x * blockDim.x + threadIdx.x;
    if (idx >= EE * 64) return;
    int edge = idx >> 6;
    int c    = (idx & 63) * 4;
    long long s = ei[edge];
    long long d = ei[EE + edge];
    const float4 hv = *(const float4*)(h + (size_t)s * HD + c);
    const float4 ev = *(const float4*)(e + (size_t)edge * HD + c);
    float* ap = aggr + (size_t)d * HD + c;
    atomicAdd(ap + 0, fmaxf(hv.x + ev.x, 0.f));
    atomicAdd(ap + 1, fmaxf(hv.y + ev.y, 0.f));
    atomicAdd(ap + 2, fmaxf(hv.z + ev.z, 0.f));
    atomicAdd(ap + 3, fmaxf(hv.w + ev.w, 0.f));
}

__global__ void pool_zero_kernel(float* gsum, float* gcnt) {
    int i = blockIdx.x * blockDim.x + threadIdx.x;
    if (i < GG * HD) gsum[i] = 0.f;
    if (i < GG)      gcnt[i] = 0.f;
}

__global__ void pool_acc_kernel(const float* __restrict__ h,
                                const long long* __restrict__ batch,
                                float* gsum, float* gcnt) {
    int idx = blockIdx.x * blockDim.x + threadIdx.x;
    if (idx >= NN * 64) return;
    int node = idx >> 6;
    int c    = (idx & 63) * 4;
    long long g = batch[node];
    const float4 hv = *(const float4*)(h + (size_t)node * HD + c);
    float* sp = gsum + (size_t)g * HD + c;
    atomicAdd(sp + 0, hv.x);
    atomicAdd(sp + 1, hv.y);
    atomicAdd(sp + 2, hv.z);
    atomicAdd(sp + 3, hv.w);
    if (c == 0) atomicAdd(gcnt + g, 1.0f);
}

// batch is sorted: first occurrence of each graph id == searchsorted result
__global__ void pool_root_kernel(const long long* __restrict__ batch,
                                 int* __restrict__ groot) {
    int i = blockIdx.x * blockDim.x + threadIdx.x;
    if (i >= NN) return;
    long long g = batch[i];
    if (i == 0 || batch[i - 1] != g) groot[(int)g] = i;
}

__global__ void pool_combine_kernel(const float* __restrict__ gsum,
                                    const float* __restrict__ gcnt,
                                    const int* __restrict__ groot,
                                    const float* __restrict__ h,
                                    const float* __restrict__ alphaPtr,
                                    float* __restrict__ pooled) {
    int i = blockIdx.x * blockDim.x + threadIdx.x;
    if (i >= GG * HD) return;
    int g = i >> 8;
    int c = i & 255;
    pooled[i] = gsum[i] / gcnt[g] +
                (*alphaPtr) * h[(size_t)groot[g] * HD + c];
}

extern "C" void kernel_launch(void* const* d_in, const int* in_sizes, int n_in,
                              void* d_out, int out_size, void* d_ws, size_t ws_size,
                              hipStream_t stream) {
    const float* x    = (const float*)d_in[0];
    const float* ea   = (const float*)d_in[1];
    const float* Wn   = (const float*)d_in[2];
    const float* bn   = (const float*)d_in[3];
    const float* We   = (const float*)d_in[4];
    const float* be   = (const float*)d_in[5];
    const float* W11  = (const float*)d_in[6];
    const float* b11  = (const float*)d_in[7];
    const float* W12  = (const float*)d_in[8];
    const float* b12  = (const float*)d_in[9];
    const float* eps1 = (const float*)d_in[10];
    const float* W21  = (const float*)d_in[11];
    const float* b21  = (const float*)d_in[12];
    const float* W22  = (const float*)d_in[13];
    const float* b22  = (const float*)d_in[14];
    const float* eps2 = (const float*)d_in[15];
    const float* Wo   = (const float*)d_in[16];
    const float* bo   = (const float*)d_in[17];
    const float* alpha= (const float*)d_in[18];
    const long long* ei    = (const long long*)d_in[19];
    const long long* batch = (const long long*)d_in[20];
    float* out = (float*)d_out;

    char* p = (char*)d_ws;
    auto carve = [&](size_t bytes) -> char* {
        char* r = p;
        p += (bytes + 255) & ~(size_t)255;
        return r;
    };
    float* e    = (float*)carve((size_t)EE * HD * 4);
    float* h0   = (float*)carve((size_t)NN * HD * 4);
    float* h1   = (float*)carve((size_t)NN * HD * 4);
    float* z    = (float*)carve((size_t)NN * HD * 4);
    float* t    = (float*)carve((size_t)NN * HD * 4);
    unsigned* WnP  = (unsigned*)carve((size_t)IN_DIM * HD * 2);
    unsigned* WeP  = (unsigned*)carve((size_t)IN_DIM * HD * 2);
    unsigned* W11P = (unsigned*)carve((size_t)HD * HD * 2);
    unsigned* W12P = (unsigned*)carve((size_t)HD * HD * 2);
    unsigned* W21P = (unsigned*)carve((size_t)HD * HD * 2);
    unsigned* W22P = (unsigned*)carve((size_t)HD * HD * 2);
    unsigned* WoP  = (unsigned*)carve((size_t)HD * HD * 2);
    float* gsum  = (float*)carve((size_t)GG * HD * 4);
    float* gcnt  = (float*)carve((size_t)GG * 4);
    int*   groot = (int*)carve((size_t)GG * 4);
    float* pooled= (float*)carve((size_t)GG * HD * 4);

    const int TB = 256;
    auto cdiv = [](long long a, long long b) { return (int)((a + b - 1) / b); };

    // ---- weight packing (fp32 -> f16 WMMA B fragments) ----
    pack_w_kernel<<<cdiv((IN_DIM / 32) * 512, TB), TB, 0, stream>>>(Wn, WnP, IN_DIM);
    pack_w_kernel<<<cdiv((IN_DIM / 32) * 512, TB), TB, 0, stream>>>(We, WeP, IN_DIM);
    pack_w_kernel<<<cdiv((HD / 32) * 512, TB), TB, 0, stream>>>(W11, W11P, HD);
    pack_w_kernel<<<cdiv((HD / 32) * 512, TB), TB, 0, stream>>>(W12, W12P, HD);
    pack_w_kernel<<<cdiv((HD / 32) * 512, TB), TB, 0, stream>>>(W21, W21P, HD);
    pack_w_kernel<<<cdiv((HD / 32) * 512, TB), TB, 0, stream>>>(W22, W22P, HD);
    pack_w_kernel<<<cdiv((HD / 32) * 512, TB), TB, 0, stream>>>(Wo, WoP, HD);

    // ---- input projections ----
    wmma_gemm_kernel<<<cdiv(NN, 32), TB, 0, stream>>>(x,  WnP, bn, nullptr, h0, NN, IN_DIM, 0);
    wmma_gemm_kernel<<<cdiv(EE, 32), TB, 0, stream>>>(ea, WeP, be, nullptr, e,  EE, IN_DIM, 0);

    // ---- GINE layer 1 ----
    scale_copy_kernel<<<cdiv((long long)NN * HD, TB), TB, 0, stream>>>(h0, eps1, z, NN * HD);
    msg_scatter_kernel<<<cdiv((long long)EE * 64, TB), TB, 0, stream>>>(h0, e, ei, z);
    wmma_gemm_kernel<<<cdiv(NN, 32), TB, 0, stream>>>(z, W11P, b11, nullptr, t,  NN, HD, F_RELU);
    wmma_gemm_kernel<<<cdiv(NN, 32), TB, 0, stream>>>(t, W12P, b12, h0,      h1, NN, HD, F_RELU | F_RESID);

    // ---- GINE layer 2 ----
    scale_copy_kernel<<<cdiv((long long)NN * HD, TB), TB, 0, stream>>>(h1, eps2, z, NN * HD);
    msg_scatter_kernel<<<cdiv((long long)EE * 64, TB), TB, 0, stream>>>(h1, e, ei, z);
    wmma_gemm_kernel<<<cdiv(NN, 32), TB, 0, stream>>>(z, W21P, b21, nullptr, t,  NN, HD, F_RELU);
    wmma_gemm_kernel<<<cdiv(NN, 32), TB, 0, stream>>>(t, W22P, b22, h1,      h0, NN, HD, F_RELU | F_RESID);

    // ---- readout: mean pool + root ----
    pool_zero_kernel<<<cdiv(GG * HD, TB), TB, 0, stream>>>(gsum, gcnt);
    pool_acc_kernel<<<cdiv((long long)NN * 64, TB), TB, 0, stream>>>(h0, batch, gsum, gcnt);
    pool_root_kernel<<<cdiv(NN, TB), TB, 0, stream>>>(batch, groot);
    pool_combine_kernel<<<cdiv(GG * HD, TB), TB, 0, stream>>>(gsum, gcnt, groot, h0, alpha, pooled);

    // ---- output projection ----
    wmma_gemm_kernel<<<cdiv(GG, 32), TB, 0, stream>>>(pooled, WoP, bo, nullptr, out, GG, HD, 0);
}
